// LocalEncoder_12403865551529
// MI455X (gfx1250) — compile-verified, matrix operand order checked by reading, not run on hardware
//
#include <hip/hip_runtime.h>
#include <math.h>

// ---------------------------------------------------------------------------
// LocalEncoder for MI455X (gfx1250): WMMA bf16 GEMMs with double-buffered
// async-to-LDS DMA staging + scalar softmax paths.
// B=2 S=1024 P=32 D=768 H=12 W=128 HASH=10000
// ---------------------------------------------------------------------------

typedef __attribute__((ext_vector_type(16))) __bf16 v16bf;
typedef __attribute__((ext_vector_type(8)))  float  v8f;

#define CB 2
#define CS 1024
#define CP 32
#define CD 768
#define CH 12
#define CW 128
#define CHASH 10000

// wave-relative LDS byte offset = low 32 bits of the generic pointer
__device__ __forceinline__ unsigned lds_off(const void* p)
{
    return (unsigned)(unsigned long long)p;
}

// CDNA5 async DMA: global -> LDS, 16 bytes per lane, tracked by ASYNCcnt.
__device__ __forceinline__ void async_load_b128(unsigned lds_byte_off, const void* gptr)
{
    asm volatile("global_load_async_to_lds_b128 %0, %1, off"
                 :: "v"(lds_byte_off), "v"(gptr) : "memory");
}

// wait until at most `n` async ops remain in flight (in-order completion)
__device__ __forceinline__ void wait_asynccnt0()
{
    asm volatile("s_wait_asynccnt 0x0" ::: "memory");
}
__device__ __forceinline__ void wait_asynccnt3()
{
    asm volatile("s_wait_asynccnt 0x3" ::: "memory");
}

// ---------------------------------------------------------------------------
// 1) byte embedding + 6 rolling-hash n-gram embeddings -> x (f32 + bf16)
// ---------------------------------------------------------------------------
__global__ __launch_bounds__(256)
void embed_kernel(const int* __restrict__ bytes_seq,
                  const float* __restrict__ byte_emb,
                  const float* __restrict__ ngram_embs,
                  float* __restrict__ xf, __bf16* __restrict__ xh)
{
    int row = blockIdx.x;            // b*S + s
    int s = row % CS, b = row / CS;
    __shared__ int hsh[6];
    if (threadIdx.x < 6) {
        int n = 3 + (int)threadIdx.x;
        unsigned long long acc = 0ULL, mlt = 1ULL;   // int64 wraparound semantics
        for (int i = 0; i < n; ++i) {
            int pos = s + i; if (pos >= CS) pos -= CS;   // jnp.roll wraps
            acc += (unsigned long long)(unsigned)bytes_seq[b * CS + pos] * mlt;
            mlt *= 257ULL;
        }
        long long h = (long long)acc % 10000LL;
        if (h < 0) h += 10000LL;                      // Python-style mod
        if (s >= CS - n + 1) h = 0;                   // valid mask
        hsh[threadIdx.x] = (int)h;
    }
    __syncthreads();
    int byte = bytes_seq[b * CS + s];
    for (int d = threadIdx.x; d < CD; d += 256) {
        float v = byte_emb[byte * CD + d];
        #pragma unroll
        for (int i = 0; i < 6; ++i)
            v += ngram_embs[((long)i * CHASH + hsh[i]) * CD + d];
        xf[(long)row * CD + d] = v;
        xh[(long)row * CD + d] = (__bf16)v;
    }
}

// ---------------------------------------------------------------------------
// 2) weight transpose + bf16 convert: in [K,N] f32 -> out [N,K] bf16.
//    32x32 LDS tile, coalesced on both sides. One-time cost per launch;
//    lets every GEMM use the pure-async [N,K] staging path.
// ---------------------------------------------------------------------------
__global__ __launch_bounds__(256)
void transpose_bf16_kernel(const float* __restrict__ in, __bf16* __restrict__ out,
                           int K, int N)
{
    __shared__ float tile[32][33];
    int k0 = blockIdx.y * 32, n0 = blockIdx.x * 32;
    int tx = threadIdx.x & 31, ty = threadIdx.x >> 5;   // 32x8
    #pragma unroll
    for (int i = 0; i < 4; ++i) {
        int k = ty + i * 8;
        tile[k][tx] = in[(long)(k0 + k) * N + n0 + tx];
    }
    __syncthreads();
    #pragma unroll
    for (int i = 0; i < 4; ++i) {
        int n = ty + i * 8;
        out[(long)(n0 + n) * K + k0 + tx] = (__bf16)tile[tx][n];
    }
}

// ---------------------------------------------------------------------------
// 3) tiled WMMA GEMM: C[M,N] = act(scale * A[M,K] @ Bt[N,K]^T + bias)
//    block tile 128x64x32, 8 waves (4x2), wave tile 32x32 = 2x2 WMMA tiles.
//    Both operands stored k-contiguous -> all staging is
//    global_load_async_to_lds_b128, double-buffered (DMA of tile t+1
//    overlaps WMMA of tile t; s_wait_asynccnt 3 retires the older batch).
//    LDS layouts fragment-ready: A [m][k], B [n][k] -> every fragment is two
//    contiguous 8-elem runs (ds_load_b128 pairs).
//    Outputs optionally f32 (Cf) and/or bf16 (Ch). Batched via grid.z.
// ---------------------------------------------------------------------------
__global__ __launch_bounds__(256)
void wmma_gemm_kernel(const __bf16* __restrict__ A, long aBatch,
                      const __bf16* __restrict__ Bt, long bBatch,
                      const float* __restrict__ bias,
                      float* __restrict__ Cf, __bf16* __restrict__ Ch, long cBatch,
                      int M, int N, int K, float scale, int relu)
{
    __shared__ __bf16 ldsA[2][128][40];   // [buf][m][k], +8 pad
    __shared__ __bf16 ldsB[2][64][40];    // [buf][n][k], +8 pad
    const int tid  = threadIdx.x;
    const int wave = tid >> 5, lane = tid & 31;
    const int wm = wave & 3, wn = wave >> 2;
    const int m0 = blockIdx.y * 128, n0 = blockIdx.x * 64;
    const long batch = blockIdx.z;
    A  += batch * aBatch;
    Bt += batch * bBatch;

    const int laneHalf = lane >> 4;       // 0/1 : lanes 16-31 carry the K+8/K+24 groups
    const int laneM    = lane & 15;
    const int koff     = laneHalf * 8;

    // staging coordinates (256 threads): 2 A rows + 1 B row of 8 bf16 each
    const int ar = tid >> 2, ac = (tid & 3) * 8;
    const __bf16* gA0 = A  + (long)(m0 + ar) * K + ac;
    const __bf16* gA1 = A  + (long)(m0 + ar + 64) * K + ac;
    const __bf16* gB  = Bt + (long)(n0 + ar) * K + ac;
    const unsigned lA0[2] = { lds_off(&ldsA[0][ar][ac]),      lds_off(&ldsA[1][ar][ac]) };
    const unsigned lA1[2] = { lds_off(&ldsA[0][ar + 64][ac]), lds_off(&ldsA[1][ar + 64][ac]) };
    const unsigned lB [2] = { lds_off(&ldsB[0][ar][ac]),      lds_off(&ldsB[1][ar][ac]) };

    v8f acc[2][2] = {};
    const int nsteps = K >> 5;

    // prologue: stage tile 0 into buffer 0
    async_load_b128(lA0[0], gA0);
    async_load_b128(lA1[0], gA1);
    async_load_b128(lB[0],  gB);

    for (int t = 0; t < nsteps; ++t) {
        const int cur = t & 1;
        if (t + 1 < nsteps) {
            // prefetch tile t+1 into the other buffer while t computes
            const long ko = (long)(t + 1) << 5;
            async_load_b128(lA0[cur ^ 1], gA0 + ko);
            async_load_b128(lA1[cur ^ 1], gA1 + ko);
            async_load_b128(lB[cur ^ 1],  gB  + ko);
            wait_asynccnt3();             // tile t landed; t+1 stays in flight
        } else {
            wait_asynccnt0();
        }
        __syncthreads();

        // fragments: two contiguous 8-elem runs each -> ds_load_b128 x2
        v16bf afr[2], bfr[2];
        #pragma unroll
        for (int i = 0; i < 2; ++i) {
            int rb = wm * 32 + i * 16 + laneM;
            #pragma unroll
            for (int j = 0; j < 8; ++j) {
                afr[i][j]     = ldsA[cur][rb][koff + j];
                afr[i][8 + j] = ldsA[cur][rb][16 + koff + j];
            }
        }
        #pragma unroll
        for (int i = 0; i < 2; ++i) {
            int nb = wn * 32 + i * 16 + laneM;
            #pragma unroll
            for (int j = 0; j < 8; ++j) {
                bfr[i][j]     = ldsB[cur][nb][koff + j];
                bfr[i][8 + j] = ldsB[cur][nb][16 + koff + j];
            }
        }
        #pragma unroll
        for (int i = 0; i < 2; ++i)
            #pragma unroll
            for (int j = 0; j < 2; ++j)
                acc[i][j] = __builtin_amdgcn_wmma_f32_16x16x32_bf16(
                    false, afr[i], false, bfr[j], (short)0, acc[i][j], false, false);
        __syncthreads();   // all waves done reading buf `cur` before it is refilled
    }

    // ---- epilogue: C/D layout is VGPR r -> row r (+8 for lanes 16-31), col = lane&15
    #pragma unroll
    for (int i = 0; i < 2; ++i) {
        #pragma unroll
        for (int j = 0; j < 2; ++j) {
            #pragma unroll
            for (int r = 0; r < 8; ++r) {
                int gm = m0 + wm * 32 + i * 16 + r + laneHalf * 8;
                int gn = n0 + wn * 32 + j * 16 + laneM;
                float v = acc[i][j][r] * scale;
                if (bias) v += bias[gn];
                if (relu) v = fmaxf(v, 0.0f);
                long idx = batch * cBatch + (long)gm * N + gn;
                if (Cf) Cf[idx] = v;
                if (Ch) Ch[idx] = (__bf16)v;
            }
        }
    }
}

// ---------------------------------------------------------------------------
// 4) causal band attention (width W+1=129), one wave32 per (b,h,q) row.
// ---------------------------------------------------------------------------
__global__ __launch_bounds__(256)
void band_attn_kernel(const float* __restrict__ qkv, __bf16* __restrict__ sah)
{
    const int dh = CD / CH;                       // 64
    __shared__ float probs[8][132];
    int wave = threadIdx.x >> 5, lane = threadIdx.x & 31;
    int row = blockIdx.x * 8 + wave;              // (b*H + h)*S + q
    int q = row % CS;
    int h = (row / CS) % CH;
    int b = row / (CS * CH);
    int lo = q - CW; if (lo < 0) lo = 0;
    int nk = q - lo + 1;                          // <= 129

    const float* qp    = qkv + ((long)(b * CS + q)) * 3 * CD + h * dh;
    const float* kbase = qkv + ((long)(b * CS)) * 3 * CD + CD + h * dh;
    const float* vbase = qkv + ((long)(b * CS)) * 3 * CD + 2 * CD + h * dh;

    float qreg[64];
    #pragma unroll
    for (int d = 0; d < 64; ++d) qreg[d] = qp[d];

    float sarr[5];
    float lmax = -1e30f;
    int cnt = 0;
    for (int i = lane; i < nk; i += 32, ++cnt) {
        const float* kp = kbase + (long)(lo + i) * 3 * CD;
        float s = 0.f;
        #pragma unroll 8
        for (int d = 0; d < 64; ++d) s += qreg[d] * kp[d];
        s *= 0.125f;                              // dh^-0.5
        sarr[cnt] = s;
        lmax = fmaxf(lmax, s);
    }
    #pragma unroll
    for (int m = 16; m > 0; m >>= 1) lmax = fmaxf(lmax, __shfl_xor(lmax, m, 32));
    float lsum = 0.f;
    cnt = 0;
    for (int i = lane; i < nk; i += 32, ++cnt) {
        float e = __expf(sarr[cnt] - lmax);
        sarr[cnt] = e;
        lsum += e;
    }
    #pragma unroll
    for (int m = 16; m > 0; m >>= 1) lsum += __shfl_xor(lsum, m, 32);
    float inv = 1.0f / lsum;
    cnt = 0;
    for (int i = lane; i < nk; i += 32, ++cnt) probs[wave][i] = sarr[cnt] * inv;
    __syncthreads();

    for (int dd = lane; dd < dh; dd += 32) {
        float a0 = 0.f;
        for (int i = 0; i < nk; ++i)
            a0 += probs[wave][i] * vbase[(long)(lo + i) * 3 * CD + dd];
        sah[((long)(b * CS + q)) * CD + h * dh + dd] = (__bf16)a0;
    }
}

// ---------------------------------------------------------------------------
// 5) residual add + layernorm (one 256-thread block per row of 768)
// ---------------------------------------------------------------------------
__global__ __launch_bounds__(256)
void ln_add_kernel(const float* __restrict__ a, const float* __restrict__ r,
                   const float* __restrict__ w, const float* __restrict__ bias,
                   float* __restrict__ outf, __bf16* __restrict__ outh)
{
    __shared__ float red[256];
    long row = blockIdx.x;
    float vals[3], lsum = 0.f;
    #pragma unroll
    for (int t = 0; t < 3; ++t) {
        int d = threadIdx.x + t * 256;
        float v = a[row * CD + d] + r[row * CD + d];
        vals[t] = v; lsum += v;
    }
    red[threadIdx.x] = lsum; __syncthreads();
    for (int s = 128; s > 0; s >>= 1) { if (threadIdx.x < s) red[threadIdx.x] += red[threadIdx.x + s]; __syncthreads(); }
    float mean = red[0] / CD; __syncthreads();
    float lss = 0.f;
    #pragma unroll
    for (int t = 0; t < 3; ++t) { float dv = vals[t] - mean; lss += dv * dv; }
    red[threadIdx.x] = lss; __syncthreads();
    for (int s = 128; s > 0; s >>= 1) { if (threadIdx.x < s) red[threadIdx.x] += red[threadIdx.x + s]; __syncthreads(); }
    float rstd = rsqrtf(red[0] / CD + 1e-5f);
    #pragma unroll
    for (int t = 0; t < 3; ++t) {
        int d = threadIdx.x + t * 256;
        float o = (vals[t] - mean) * rstd * w[d] + bias[d];
        if (outf) outf[row * CD + d] = o;
        if (outh) outh[row * CD + d] = (__bf16)o;
    }
}

// ---------------------------------------------------------------------------
// 6) patch-pool cross attention + final LN, fused. grid = (B*P, S/16).
// ---------------------------------------------------------------------------
__global__ __launch_bounds__(256)
void cross_attn_kernel(const int* __restrict__ patch_indices,
                       const float* __restrict__ sc,
                       const float* __restrict__ v,
                       const float* __restrict__ lnf_w, const float* __restrict__ lnf_b,
                       float* __restrict__ out)
{
    __shared__ float probs[CS];
    __shared__ float red[256];
    int bp = blockIdx.x, b = bp / CP, p = bp % CP;
    int start = patch_indices[bp];
    int end   = (p == CP - 1) ? CS : patch_indices[bp + 1];
    int len   = end - start;
    int q0 = blockIdx.y * 16;

    for (int q = q0; q < q0 + 16; ++q) {
        const float* scr = sc + ((long)b * CS + q) * CS + start;
        float lmax = -1e30f;
        for (int k = threadIdx.x; k < len; k += 256) lmax = fmaxf(lmax, scr[k]);
        red[threadIdx.x] = lmax; __syncthreads();
        for (int s = 128; s > 0; s >>= 1) { if (threadIdx.x < s) red[threadIdx.x] = fmaxf(red[threadIdx.x], red[threadIdx.x + s]); __syncthreads(); }
        float gmax = red[0]; __syncthreads();

        float lsum = 0.f;
        for (int k = threadIdx.x; k < len; k += 256) { float e = __expf(scr[k] - gmax); probs[k] = e; lsum += e; }
        red[threadIdx.x] = lsum; __syncthreads();
        for (int s = 128; s > 0; s >>= 1) { if (threadIdx.x < s) red[threadIdx.x] += red[threadIdx.x + s]; __syncthreads(); }
        float inv = 1.0f / red[0]; __syncthreads();

        float od[3], lsum2 = 0.f;
        #pragma unroll
        for (int t = 0; t < 3; ++t) {
            int d = threadIdx.x + t * 256;
            const float* vp = v + ((long)b * CS + start) * CD + d;
            float accv = 0.f;
            for (int k = 0; k < len; ++k) accv += probs[k] * vp[(long)k * CD];
            od[t] = accv * inv;
            lsum2 += od[t];
        }
        red[threadIdx.x] = lsum2; __syncthreads();
        for (int s = 128; s > 0; s >>= 1) { if (threadIdx.x < s) red[threadIdx.x] += red[threadIdx.x + s]; __syncthreads(); }
        float mean = red[0] / CD; __syncthreads();
        float lss = 0.f;
        #pragma unroll
        for (int t = 0; t < 3; ++t) { float dv = od[t] - mean; lss += dv * dv; }
        red[threadIdx.x] = lss; __syncthreads();
        for (int s = 128; s > 0; s >>= 1) { if (threadIdx.x < s) red[threadIdx.x] += red[threadIdx.x + s]; __syncthreads(); }
        float rstd = rsqrtf(red[0] / CD + 1e-5f); __syncthreads();

        long obase = ((long)bp * CS + q) * CD;
        #pragma unroll
        for (int t = 0; t < 3; ++t) {
            int d = threadIdx.x + t * 256;
            out[obase + d] = (od[t] - mean) * rstd * lnf_w[d] + lnf_b[d];
        }
    }
}

// ---------------------------------------------------------------------------
// orchestration
// ---------------------------------------------------------------------------
extern "C" void kernel_launch(void* const* d_in, const int* in_sizes, int n_in,
                              void* d_out, int out_size, void* d_ws, size_t ws_size,
                              hipStream_t stream)
{
    (void)in_sizes; (void)n_in; (void)out_size; (void)ws_size;
    const int  B = CB, S = CS, P = CP, D = CD, H = CH;
    const long M  = (long)B * S;          // 2048 rows
    const int  D3 = 3 * D, D4 = 4 * D;

    const int*   bytes_seq  = (const int*)  d_in[0];
    const int*   patch_idx  = (const int*)  d_in[1];
    const float* byte_emb   = (const float*)d_in[2];
    const float* ngram_embs = (const float*)d_in[3];
    const float* in_proj_w  = (const float*)d_in[4];
    const float* in_proj_b  = (const float*)d_in[5];
    const float* out_w      = (const float*)d_in[6];
    const float* out_b      = (const float*)d_in[7];
    const float* ln1_w      = (const float*)d_in[8];
    const float* ln1_b      = (const float*)d_in[9];
    const float* ff1_w      = (const float*)d_in[10];
    const float* ff1_b      = (const float*)d_in[11];
    const float* ff2_w      = (const float*)d_in[12];
    const float* ff2_b      = (const float*)d_in[13];
    const float* ln2_w      = (const float*)d_in[14];
    const float* ln2_b      = (const float*)d_in[15];
    const float* cq_w       = (const float*)d_in[16];
    const float* cq_b       = (const float*)d_in[17];
    const float* ck_w       = (const float*)d_in[18];
    const float* ck_b       = (const float*)d_in[19];
    const float* cv_w       = (const float*)d_in[20];
    const float* cv_b       = (const float*)d_in[21];
    const float* lnf_w      = (const float*)d_in[22];
    const float* lnf_b      = (const float*)d_in[23];

    // bump allocator over workspace (~110 MB used)
    char* ws = (char*)d_ws;
    size_t off = 0;
    auto alloc = [&](size_t bytes) -> void* {
        void* pp = ws + off;
        off = (off + bytes + 255) & ~(size_t)255;
        return pp;
    };
    float*  xf     = (float*) alloc(M * D * 4);
    __bf16* xh     = (__bf16*)alloc(M * D * 2);
    __bf16* w_inpT = (__bf16*)alloc((long)D3 * D * 2);  // [N,K] transposed weights
    __bf16* w_outT = (__bf16*)alloc((long)D * D * 2);
    __bf16* w_ff1T = (__bf16*)alloc((long)D4 * D * 2);
    __bf16* w_ff2T = (__bf16*)alloc((long)D * D4 * 2);
    __bf16* w_cqT  = (__bf16*)alloc((long)D * D * 2);
    __bf16* w_ckT  = (__bf16*)alloc((long)D * D * 2);
    __bf16* w_cvT  = (__bf16*)alloc((long)D * D * 2);
    float*  qkv    = (float*) alloc(M * D3 * 4);
    __bf16* sah    = (__bf16*)alloc(M * D * 2);
    float*  yf     = (float*) alloc(M * D * 4);    // reused: sa_o, then f2
    float*  x1f    = (float*) alloc(M * D * 4);
    __bf16* x1h    = (__bf16*)alloc(M * D * 2);
    __bf16* f1h    = (__bf16*)alloc(M * D4 * 2);
    __bf16* x2h    = (__bf16*)alloc(M * D * 2);
    __bf16* qh     = (__bf16*)alloc(M * D * 2);
    __bf16* kh     = (__bf16*)alloc(M * D * 2);
    float*  vf     = (float*) alloc(M * D * 4);
    float*  scf    = (float*) alloc((long)B * S * S * 4);

    // 1) embeddings + n-gram hashes
    embed_kernel<<<(int)M, 256, 0, stream>>>(bytes_seq, byte_emb, ngram_embs, xf, xh);

    // 2) weight transpose+convert: [K,N] f32 -> [N,K] bf16
    auto cvtT = [&](const float* src, __bf16* dst, int K, int N) {
        transpose_bf16_kernel<<<dim3(N / 32, K / 32), 256, 0, stream>>>(src, dst, K, N);
    };
    cvtT(in_proj_w, w_inpT, D, D3);
    cvtT(out_w,     w_outT, D, D);
    cvtT(ff1_w,     w_ff1T, D, D4);
    cvtT(ff2_w,     w_ff2T, D4, D);
    cvtT(cq_w,      w_cqT,  D, D);
    cvtT(ck_w,      w_ckT,  D, D);
    cvtT(cv_w,      w_cvT,  D, D);

    // 3) qkv = x @ in_proj_w + b
    wmma_gemm_kernel<<<dim3(D3 / 64, (unsigned)(M / 128), 1), 256, 0, stream>>>(
        xh, 0, w_inpT, 0, in_proj_b, qkv, nullptr, 0, (int)M, D3, D, 1.0f, 0);

    // 4) causal band self-attention -> sah (bf16)
    band_attn_kernel<<<(B * H * S) / 8, 256, 0, stream>>>(qkv, sah);

    // 5) out projection -> yf
    wmma_gemm_kernel<<<dim3(D / 64, (unsigned)(M / 128), 1), 256, 0, stream>>>(
        sah, 0, w_outT, 0, out_b, yf, nullptr, 0, (int)M, D, D, 1.0f, 0);

    // 6) x1 = LN(x + sa_o)
    ln_add_kernel<<<(int)M, 256, 0, stream>>>(xf, yf, ln1_w, ln1_b, x1f, x1h);

    // 7) f1 = relu(x1 @ ff1_w + b)  (bf16 only)
    wmma_gemm_kernel<<<dim3(D4 / 64, (unsigned)(M / 128), 1), 256, 0, stream>>>(
        x1h, 0, w_ff1T, 0, ff1_b, nullptr, f1h, 0, (int)M, D4, D, 1.0f, 1);

    // 8) f2 = f1 @ ff2_w + b -> yf (reuse)
    wmma_gemm_kernel<<<dim3(D / 64, (unsigned)(M / 128), 1), 256, 0, stream>>>(
        f1h, 0, w_ff2T, 0, ff2_b, yf, nullptr, 0, (int)M, D, D4, 1.0f, 0);

    // 9) x2 = LN(x1 + f2)  (bf16 only)
    ln_add_kernel<<<(int)M, 256, 0, stream>>>(x1f, yf, ln2_w, ln2_b, nullptr, x2h);

    // 10) cross-attention projections
    wmma_gemm_kernel<<<dim3(D / 64, (unsigned)(M / 128), 1), 256, 0, stream>>>(
        x2h, 0, w_cqT, 0, cq_b, nullptr, qh, 0, (int)M, D, D, 1.0f, 0);
    wmma_gemm_kernel<<<dim3(D / 64, (unsigned)(M / 128), 1), 256, 0, stream>>>(
        x2h, 0, w_ckT, 0, ck_b, nullptr, kh, 0, (int)M, D, D, 1.0f, 0);
    wmma_gemm_kernel<<<dim3(D / 64, (unsigned)(M / 128), 1), 256, 0, stream>>>(
        x2h, 0, w_cvT, 0, cv_b, vf, nullptr, 0, (int)M, D, D, 1.0f, 0);

    // 11) sc[b] = q[b] @ k[b]^T * D^-0.5 (batched; kh rows are k-contiguous already)
    wmma_gemm_kernel<<<dim3(S / 64, S / 128, B), 256, 0, stream>>>(
        qh, (long)S * D, kh, (long)S * D, nullptr, scf, nullptr, (long)S * S,
        S, S, D, 1.0f / sqrtf((float)D), 0);

    // 12) patch-pool softmax + weighted sum + final LN -> d_out
    cross_attn_kernel<<<dim3(B * P, S / 16), 256, 0, stream>>>(
        patch_idx, scf, vf, lnf_w, lnf_b, (float*)d_out);
}
